// FusedLinearCrossEntropyLoss_78305843741014
// MI455X (gfx1250) — matrix operand order repaired and checked by reading, not compiled, not run on hardware
//
#include <hip/hip_runtime.h>
#include <math.h>

// ---------------------------------------------------------------------------
// Fused Linear + Cross-Entropy for MI455X (gfx1250, wave32, WMMA bf16)
//   x[4096,1024] fp32, W[32000,1024] fp32, bias[32000] fp32, target[4096] int
// Strategy: pre-convert x/W to bf16 once (workspace permitting), then a
// WMMA bf16 GEMM that never materializes logits: online logsumexp + target
// gather fused in the epilogue, deterministic tree reduction at the end.
// Tile staging uses gfx1250 async global->LDS copies when available.
// ---------------------------------------------------------------------------

typedef __bf16 bf16;
typedef __attribute__((ext_vector_type(16))) __bf16 v16bf;
typedef __attribute__((ext_vector_type(8)))  __bf16 v8bf;
typedef __attribute__((ext_vector_type(4)))  __bf16 v4bf;
typedef __attribute__((ext_vector_type(8)))  float  v8f;
typedef int v4i_vs __attribute__((vector_size(16)));   // matches async-LDS builtin param

#define BT       4096
#define H        1024
#define NV       32000
#define M_TILE   64
#define N_TILE   128
#define K_CHUNK  128
#define SA       136   // LDS row stride (halves): 68 dwords -> bank step 4, conflict-free
#define SB       136
#define V_SPLITS 5
#define CHUNKS   50    // 50 * 128 * 5 = 32000
#define SLOTS    (V_SPLITS * 2)
#define IGNORE_INDEX (-100)

// ---- gfx1250 async global->LDS copy (guarded; falls back to load+ds_store)
#if defined(__has_builtin)
#if __has_builtin(__builtin_amdgcn_global_load_async_to_lds_b128) && \
    __has_builtin(__builtin_amdgcn_s_wait_asynccnt)
#define USE_ASYNC_LDS 1
#endif
#endif

__device__ __forceinline__ void g2l_b128(const void* g, void* l) {
#ifdef USE_ASYNC_LDS
    __builtin_amdgcn_global_load_async_to_lds_b128(
        (__attribute__((address_space(1))) v4i_vs*)g,
        (__attribute__((address_space(3))) v4i_vs*)l, 0, 0);
#else
    *(v8bf*)l = *(const v8bf*)g;
#endif
}
__device__ __forceinline__ void async_fence() {
#ifdef USE_ASYNC_LDS
    __builtin_amdgcn_s_wait_asynccnt(0);
#endif
}

// ---- fp32 -> bf16 bulk conversion (one pass; makes main-kernel traffic bf16)
__global__ __launch_bounds__(256)
void cvt_f32_to_bf16(const float* __restrict__ in, bf16* __restrict__ out)
{
    const size_t i  = ((size_t)blockIdx.x * 256 + threadIdx.x) * 8;
    const float4* in4 = (const float4*)in;
    float4 a = in4[i / 4];
    float4 b = in4[i / 4 + 1];
    v8bf o;
    o[0] = (bf16)a.x; o[1] = (bf16)a.y; o[2] = (bf16)a.z; o[3] = (bf16)a.w;
    o[4] = (bf16)b.x; o[5] = (bf16)b.y; o[6] = (bf16)b.z; o[7] = (bf16)b.w;
    *(v8bf*)&out[i] = o;
}

// XB: x available as bf16; WB: W available as bf16
template <bool XB, bool WB>
__global__ __launch_bounds__(256)
void flce_main(const float* __restrict__ x,   const bf16* __restrict__ xbf,
               const int*   __restrict__ tgt,
               const float* __restrict__ w,   const bf16* __restrict__ wbf,
               const float* __restrict__ bias,
               float*       __restrict__ partials)
{
    __shared__ __align__(16) bf16 sA[M_TILE * SA];   // 17408 B
    __shared__ __align__(16) bf16 sB[N_TILE * SB];   // 34816 B

    const int tid  = threadIdx.x;
    const int lane = tid & 31;
    const int wave = tid >> 5;
    const int wm   = wave & 3;    // M slice (16 rows)
    const int wn   = wave >> 2;   // N half (64 cols)
    const int row_base = blockIdx.x * M_TILE;
    const int vs   = blockIdx.y;
    const int l16  = lane & 15;
    const int hi   = lane >> 4;

    int tg[8];
#pragma unroll
    for (int j = 0; j < 8; ++j)
        tg[j] = tgt[row_base + wm * 16 + j + (hi << 3)];

    float m_run[8], s_run[8], xy[8];
#pragma unroll
    for (int j = 0; j < 8; ++j) { m_run[j] = -INFINITY; s_run[j] = 0.f; xy[j] = 0.f; }

    const float4* x4 = (const float4*)x;
    const float4* w4 = (const float4*)w;
    const v8f zero8 = {0.f, 0.f, 0.f, 0.f, 0.f, 0.f, 0.f, 0.f};

    for (int ci = 0; ci < CHUNKS; ++ci) {
        const int vbase = (vs * CHUNKS + ci) * N_TILE;

        if (ci + 1 < CHUNKS) {   // hint-prefetch next V-chunk's weights
            const int vn = (vs * CHUNKS + ci + 1) * N_TILE;
            if constexpr (WB)
                __builtin_prefetch(&wbf[(size_t)(vn + (tid >> 1)) * H + (size_t)(tid & 1) * 512], 0, 1);
            else
                __builtin_prefetch(&w[(size_t)(vn + (tid >> 1)) * H + (size_t)(tid & 1) * 512], 0, 1);
        }

        v8f acc[4];
#pragma unroll
        for (int n = 0; n < 4; ++n) acc[n] = zero8;

        for (int kc = 0; kc < H / K_CHUNK; ++kc) {
            const int kbase = kc * K_CHUNK;
            __syncthreads();

            // ---- stage A tile: 64 rows x 128 K (bf16 in LDS)
            if constexpr (XB) {
#pragma unroll
                for (int it = 0; it < 4; ++it) {            // 1024 b128 chunks
                    int idx = tid + it * 256;
                    int r   = idx >> 4;                     // 16 chunks per row
                    int kh  = (idx & 15) << 3;              // 8 halves per chunk
                    g2l_b128(&xbf[(size_t)(row_base + r) * H + kbase + kh],
                             &sA[r * SA + kh]);
                }
            } else {
#pragma unroll
                for (int it = 0; it < 8; ++it) {            // 2048 float4s
                    int idx4 = tid + it * 256;
                    int r    = idx4 >> 5;
                    int kq   = (idx4 & 31) << 2;
                    float4 v = x4[(size_t)(row_base + r) * (H / 4) + ((kbase + kq) >> 2)];
                    v4bf b;
                    b[0] = (bf16)v.x; b[1] = (bf16)v.y; b[2] = (bf16)v.z; b[3] = (bf16)v.w;
                    *(v4bf*)&sA[r * SA + kq] = b;
                }
            }
            // ---- stage B tile: 128 cols x 128 K
            if constexpr (WB) {
#pragma unroll
                for (int it = 0; it < 8; ++it) {            // 2048 b128 chunks
                    int idx = tid + it * 256;
                    int c   = idx >> 4;
                    int kh  = (idx & 15) << 3;
                    g2l_b128(&wbf[(size_t)(vbase + c) * H + kbase + kh],
                             &sB[c * SB + kh]);
                }
            } else {
#pragma unroll
                for (int it = 0; it < 16; ++it) {           // 4096 float4s
                    int idx4 = tid + it * 256;
                    int c    = idx4 >> 5;
                    int kq   = (idx4 & 31) << 2;
                    float4 v = w4[(size_t)(vbase + c) * (H / 4) + ((kbase + kq) >> 2)];
                    v4bf b;
                    b[0] = (bf16)v.x; b[1] = (bf16)v.y; b[2] = (bf16)v.z; b[3] = (bf16)v.w;
                    *(v4bf*)&sB[c * SB + kq] = b;
                }
            }
            if constexpr (XB || WB) async_fence();          // s_wait_asynccnt 0
            __syncthreads();

#pragma unroll
            for (int kb = 0; kb < K_CHUNK; kb += 32) {
                // A fragment (16x32): lanes 0-15 hold K 0-7 & 16-23; 16-31 hold 8-15 & 24-31
                const int arow  = wm * 16 + l16;
                const int akoff = kb + (hi ? 8 : 0);
                v8bf a0 = *(const v8bf*)&sA[arow * SA + akoff];
                v8bf a1 = *(const v8bf*)&sA[arow * SA + akoff + 16];
                v16bf a = __builtin_shufflevector(a0, a1,
                    0, 1, 2, 3, 4, 5, 6, 7, 8, 9, 10, 11, 12, 13, 14, 15);
#pragma unroll
                for (int n = 0; n < 4; ++n) {
                    // B fragment (32x16): column = lane%16; lanes 0-15 hold K 0-7 & 16-23; 16-31 hold 8-15 & 24-31
                    const int bcol  = wn * 64 + n * 16 + l16;
                    const int bkoff = kb + (hi ? 8 : 0);
                    v8bf b0 = *(const v8bf*)&sB[bcol * SB + bkoff];
                    v8bf b1 = *(const v8bf*)&sB[bcol * SB + bkoff + 16];
                    v16bf b = __builtin_shufflevector(b0, b1,
                        0, 1, 2, 3, 4, 5, 6, 7, 8, 9, 10, 11, 12, 13, 14, 15);
                    acc[n] = __builtin_amdgcn_wmma_f32_16x16x32_bf16(
                        false, a, false, b, (short)0, acc[n], false, false);
                }
            }
        }

        // ---- epilogue: bias, target gather, online logsumexp
        float bval[4];
#pragma unroll
        for (int n = 0; n < 4; ++n) bval[n] = bias[vbase + wn * 64 + n * 16 + l16];

#pragma unroll
        for (int j = 0; j < 8; ++j) {
            float v0 = acc[0][j] + bval[0];
            float v1 = acc[1][j] + bval[1];
            float v2 = acc[2][j] + bval[2];
            float v3 = acc[3][j] + bval[3];
            const int cg = vbase + wn * 64 + l16;
            if (tg[j] == cg     ) xy[j] = v0;
            if (tg[j] == cg + 16) xy[j] = v1;
            if (tg[j] == cg + 32) xy[j] = v2;
            if (tg[j] == cg + 48) xy[j] = v3;
            float cm = fmaxf(fmaxf(v0, v1), fmaxf(v2, v3));
            float cs = __expf(v0 - cm) + __expf(v1 - cm) + __expf(v2 - cm) + __expf(v3 - cm);
            float nm = fmaxf(m_run[j], cm);
            s_run[j] = s_run[j] * __expf(m_run[j] - nm) + cs * __expf(cm - nm);
            m_run[j] = nm;
        }
    }

    // reduce (m, s, xy) across the 16 lanes of each half-wave
#pragma unroll
    for (int j = 0; j < 8; ++j) {
        float m = m_run[j], s = s_run[j], g = xy[j];
#pragma unroll
        for (int mask = 1; mask <= 8; mask <<= 1) {
            float om = __shfl_xor(m, mask, 32);
            float os = __shfl_xor(s, mask, 32);
            float og = __shfl_xor(g, mask, 32);
            float nm = fmaxf(m, om);
            s = s * __expf(m - nm) + os * __expf(om - nm);
            m = nm;
            g += og;
        }
        m_run[j] = m; s_run[j] = s; xy[j] = g;
    }

    if (l16 == 0) {
        const int slot = vs * 2 + wn;
#pragma unroll
        for (int j = 0; j < 8; ++j) {
            int row = row_base + wm * 16 + j + (hi << 3);
            float* p = partials + ((size_t)row * SLOTS + slot) * 3;
            p[0] = m_run[j]; p[1] = s_run[j]; p[2] = xy[j];
        }
    }
}

__global__ __launch_bounds__(256)
void flce_reduce(const float* __restrict__ partials,
                 const int*   __restrict__ tgt,
                 float* __restrict__ bsum, float* __restrict__ bcnt)
{
    __shared__ float ssum[256];
    __shared__ float scnt[256];
    const int tid = threadIdx.x;
    const int row = blockIdx.x * 256 + tid;

    float m = -INFINITY, s = 0.f, g = 0.f;
#pragma unroll
    for (int k = 0; k < SLOTS; ++k) {
        const float* p = partials + ((size_t)row * SLOTS + k) * 3;
        float pm = p[0], ps = p[1];
        g += p[2];
        float nm = fmaxf(m, pm);
        s = s * __expf(m - nm) + ps * __expf(pm - nm);
        m = nm;
    }
    const float lse = m + __logf(s);
    const bool valid = (tgt[row] != IGNORE_INDEX);
    ssum[tid] = valid ? (lse - g) : 0.f;
    scnt[tid] = valid ? 1.f : 0.f;
    __syncthreads();
    for (int off = 128; off > 0; off >>= 1) {
        if (tid < off) { ssum[tid] += ssum[tid + off]; scnt[tid] += scnt[tid + off]; }
        __syncthreads();
    }
    if (tid == 0) { bsum[blockIdx.x] = ssum[0]; bcnt[blockIdx.x] = scnt[0]; }
}

__global__ void flce_final(const float* __restrict__ bsum,
                           const float* __restrict__ bcnt,
                           float* __restrict__ out)
{
    float s = 0.f, c = 0.f;
#pragma unroll
    for (int i = 0; i < BT / 256; ++i) { s += bsum[i]; c += bcnt[i]; }
    out[0] = s / c;
}

extern "C" void kernel_launch(void* const* d_in, const int* in_sizes, int n_in,
                              void* d_out, int out_size, void* d_ws, size_t ws_size,
                              hipStream_t stream)
{
    const float* x      = (const float*)d_in[0];
    const int*   target = (const int*)  d_in[1];
    const float* weight = (const float*)d_in[2];
    const float* bias   = (const float*)d_in[3];
    float*       out    = (float*)d_out;

    char* ws = (char*)d_ws;
    // workspace layout (bytes)
    const size_t OFF_PART = 0;                                  // BT*SLOTS*3 f32 + 2*16 f32
    const size_t OFF_XBF  = 512 * 1024;                         // 8 MB x bf16
    const size_t OFF_WBF  = OFF_XBF + (size_t)BT * H * 2;       // 64 MB W bf16
    const size_t END_WBF  = OFF_WBF + (size_t)NV * H * 2;

    float* partials = (float*)(ws + OFF_PART);
    float* bsum     = partials + (size_t)BT * SLOTS * 3;
    float* bcnt     = bsum + (BT / 256);
    bf16*  xbf      = (bf16*)(ws + OFF_XBF);
    bf16*  wbf      = (bf16*)(ws + OFF_WBF);

    const bool haveX = ws_size >= OFF_WBF;   // room for x bf16
    const bool haveW = ws_size >= END_WBF;   // room for W bf16 too

    if (haveX)
        cvt_f32_to_bf16<<<(BT * H) / 2048, 256, 0, stream>>>(x, xbf);
    if (haveW)
        cvt_f32_to_bf16<<<((size_t)NV * H) / 2048, 256, 0, stream>>>(weight, wbf);

    dim3 grid(BT / M_TILE, V_SPLITS);   // 64 x 5 = 320 workgroups
    if (haveW && haveX)
        flce_main<true,  true ><<<grid, 256, 0, stream>>>(x, xbf, target, weight, wbf, bias, partials);
    else if (haveX)
        flce_main<true,  false><<<grid, 256, 0, stream>>>(x, xbf, target, weight, wbf, bias, partials);
    else
        flce_main<false, false><<<grid, 256, 0, stream>>>(x, xbf, target, weight, wbf, bias, partials);

    flce_reduce<<<BT / 256, 256, 0, stream>>>(partials, target, bsum, bcnt);
    flce_final<<<1, 1, 0, stream>>>(bsum, bcnt, out);
}
